// FastMIPL_22728966930552
// MI455X (gfx1250) — matrix operand order repaired and checked by reading, not compiled, not run on hardware
//
#include <hip/hip_runtime.h>
#include <hip/hip_bf16.h>

// FastMIPL pipeline for gfx1250 (MI455X): wave32, WMMA f16, TDM tile staging.
// N=262144 instances, Q=256 feats, P*S=80 heads, B=4096 bags.

typedef _Float16 half_t;
typedef __attribute__((ext_vector_type(16))) _Float16 v16h;
typedef __attribute__((ext_vector_type(8)))  _Float16 v8h;
typedef __attribute__((ext_vector_type(8)))  float    v8f;
typedef __attribute__((ext_vector_type(4)))  unsigned v4ui;
typedef __attribute__((ext_vector_type(8)))  int      v8i;
typedef __attribute__((ext_vector_type(4)))  int      v4i;

#define NN 262144
#define QQ 256
#define PS 80
#define CC 160   // [beta_u | beta_z/b] concatenated columns
#define NB 4096

#define XROW 260      // x LDS row stride in dwords  (256 + 4 pad, TDM-padded)
#define BROW 264      // B LDS row stride in halves  (256 + 8 pad, TDM-padded)
#define XBYTES (64 * XROW * 4)            // 66,560
#define BBYTES (CC * BROW * 2)            // 84,480
#define SMEM_BYTES (XBYTES + BBYTES)      // 151,040  (< 320KB/WGP, dynamic LDS)

union V16U { v16h v; v8h h[2]; };

// ---------------------------------------------------------------------------
// K1: b = sqrt(mean_q beta_z^2); build K-major f16 operand BhT[160][256]
//     columns 0..79  = beta_u,  columns 80..159 = beta_z * (1/b)  (== eta)
// ---------------------------------------------------------------------------
__global__ __launch_bounds__(256) void prep_kernel(const float* __restrict__ bu,
                                                   const float* __restrict__ bz,
                                                   half_t* __restrict__ BhT,
                                                   float* __restrict__ bvec) {
    __shared__ float inv[PS];
    const int t = threadIdx.x;
    if (t < PS) {
        float ss = 0.f;
        for (int q = 0; q < QQ; ++q) {
            float v = bz[q * PS + t];
            ss += v * v;
        }
        float b = sqrtf(ss * (1.0f / (float)QQ));
        bvec[t] = b;
        inv[t] = 1.0f / b;
    }
    __syncthreads();
    for (int idx = t; idx < CC * QQ; idx += 256) {
        const int c = idx >> 8;      // output column 0..159
        const int q = idx & 255;     // K index
        float v = (c < PS) ? bu[q * PS + c]
                           : bz[q * PS + (c - PS)] * inv[c - PS];
        BhT[c * QQ + q] = (half_t)v;   // K-major: contiguous K per column
    }
}

// ---------------------------------------------------------------------------
// TDM issue helper (ISA ch.8 D# layout; 5-arg builtin on clang-22/ROCm 7.2,
// 6-arg on clang-23/amdgpu-toolchain).
// ---------------------------------------------------------------------------
__device__ __forceinline__ void tdm_load(v4ui g0, v8i g1) {
    const v4i z4 = {0, 0, 0, 0};
#if defined(__clang_major__) && (__clang_major__ >= 23)
    const v8i z8 = {0, 0, 0, 0, 0, 0, 0, 0};
    __builtin_amdgcn_tensor_load_to_lds(g0, g1, z4, z4, z8, 0);
#else
    __builtin_amdgcn_tensor_load_to_lds(g0, g1, z4, z4, 0);
#endif
}

// ---------------------------------------------------------------------------
// K2: fused GEMM  WT[N][160] = x[N][256] * [beta_u | eta][256][160]
//     64-row slab per workgroup, 8 waves: mtile = wid&3, 5 ntiles per wave.
//     Both operands staged into LDS by the Tensor Data Mover:
//       desc X: 64x256 f32 tile of x, row-padded to 260 dwords (bank-safe)
//       desc B: whole 160x256 f16 K-major operand, row-padded to 132 dwords
//     A fragments: ds_load_b128 x4 + f32->f16 convert (RTE, same as round 1).
//     B fragments: two contiguous 16B ds loads per (ntile, kstep).
// ---------------------------------------------------------------------------
__global__ __launch_bounds__(256) void gemm_kernel(const float* __restrict__ x,
                                                   const half_t* __restrict__ BhT,
                                                   float* __restrict__ WT) {
    extern __shared__ char smem[];
    float*  xs = (float*)smem;               // [64][XROW]
    half_t* bs = (half_t*)(smem + XBYTES);   // [CC][BROW]

    const int tid = threadIdx.x;
    const int rowBase = blockIdx.x * 64;

    if (tid == 0) {
        // --- descriptor for x tile: data_size=4B(code 2), pad_enable,
        //     pad_interval=256 dwords(code 7), pad_amount=4 dwords(code 3)
        const unsigned long long gx =
            (unsigned long long)(uintptr_t)x + (unsigned long long)rowBase * QQ * 4ull;
        v4ui g0x = { 1u,                                  // count=1 (valid D#)
                     0u,                                  // lds_addr = 0
                     (unsigned)(gx & 0xFFFFFFFFu),
                     (unsigned)((gx >> 32) & 0x01FFFFFFu) | 0x80000000u };  // type=2
        v8i g1x = { (int)0x07D20000,      // pad(3,7)+pad_en+data_size=2
                    (int)0x01000000,      // tensor_dim0 = 256 (low16<<16)
                    0,                    // tensor_dim0 hi / tensor_dim1 lo16(=0)
                    (int)0x01000004,      // tensor_dim1 hi (=4) | tile_dim0=256
                    64,                   // tile_dim1 = 64 rows
                    256,                  // tensor_dim0_stride = 256
                    0, 0 };
        tdm_load(g0x, g1x);

        // --- descriptor for B operand: data_size=2B(code 1), pad_enable,
        //     pad_interval=128 dwords(code 6), pad_amount=4 dwords(code 3)
        const unsigned long long gb = (unsigned long long)(uintptr_t)BhT;
        v4ui g0b = { 1u,
                     (unsigned)XBYTES,    // lds_addr after x region
                     (unsigned)(gb & 0xFFFFFFFFu),
                     (unsigned)((gb >> 32) & 0x01FFFFFFu) | 0x80000000u };
        v8i g1b = { (int)0x07910000,      // pad(3,6)+pad_en+data_size=1
                    (int)0x01000000,      // tensor_dim0 = 256
                    (int)0x00A00000,      // tensor_dim1 = 160 (low16<<16)
                    (int)0x01000000,      // tile_dim0 = 256
                    160,                  // tile_dim1 = 160 rows
                    256,                  // tensor_dim0_stride = 256
                    0, 0 };
        tdm_load(g0b, g1b);

        __builtin_amdgcn_s_wait_tensorcnt(0);
    }
    __syncthreads();

    const int wid  = tid >> 5;
    const int lane = tid & 31;
    const int mtile  = wid & 3;          // 4 M-tiles of 16 rows
    const int ntile0 = (wid >> 2) * 5;   // 2 wave-groups x 5 N-tiles = 10
    const int hs = lane >> 4;            // half-wave select
    const int lm = lane & 15;
    const int arow = mtile * 16 + lm;

    v8f acc[5] = {};

    for (int ks = 0; ks < 8; ++ks) {     // K = 8 * 32
        const int kbase = ks * 32;
        // A fragment (16x32 f16): lane lm = row M; halves 0..7 -> K=hs*8+0..7,
        // halves 8..15 -> K=16+hs*8+0..7 (ISA 16-bit A layout). Convert f32->f16.
        const float4* pa0 = (const float4*)&xs[arow * XROW + kbase + hs * 8];
        const float4* pa1 = (const float4*)&xs[arow * XROW + kbase + 16 + hs * 8];
        const float4 f0 = pa0[0], f1 = pa0[1];
        const float4 f2 = pa1[0], f3 = pa1[1];
        V16U a;
        a.v[0]  = (half_t)f0.x; a.v[1]  = (half_t)f0.y;
        a.v[2]  = (half_t)f0.z; a.v[3]  = (half_t)f0.w;
        a.v[4]  = (half_t)f1.x; a.v[5]  = (half_t)f1.y;
        a.v[6]  = (half_t)f1.z; a.v[7]  = (half_t)f1.w;
        a.v[8]  = (half_t)f2.x; a.v[9]  = (half_t)f2.y;
        a.v[10] = (half_t)f2.z; a.v[11] = (half_t)f2.w;
        a.v[12] = (half_t)f3.x; a.v[13] = (half_t)f3.y;
        a.v[14] = (half_t)f3.z; a.v[15] = (half_t)f3.w;
#pragma unroll
        for (int t = 0; t < 5; ++t) {
            // B fragment (32x16 f16): lane lm = column N; halves 0..7 -> K=hs*8+0..7,
            // halves 8..15 -> K=16+hs*8+0..7 (same K order as A)
            const int c = (ntile0 + t) * 16 + lm;
            const half_t* bp = &bs[c * BROW + kbase + hs * 8];
            V16U b;
            b.h[0] = *(const v8h*)bp;
            b.h[1] = *(const v8h*)(bp + 16);
            acc[t] = __builtin_amdgcn_wmma_f32_16x16x32_f16(
                false, a.v, false, b.v, (short)0, acc[t], false, false);
        }
    }

    // C/D layout: VGPR r holds (m = hs*8 + r, n = lm)
#pragma unroll
    for (int t = 0; t < 5; ++t) {
        const int col = (ntile0 + t) * 16 + lm;
#pragma unroll
        for (int r = 0; r < 8; ++r) {
            const int row = rowBase + mtile * 16 + hs * 8 + r;
            WT[(size_t)row * CC + col] = acc[t][r];
        }
    }
}

// ---------------------------------------------------------------------------
// K3: bag boundaries via binary search over sorted seg_ids
// ---------------------------------------------------------------------------
__global__ void bounds_kernel(const int* __restrict__ seg,
                              int* __restrict__ starts) {
    const int b = blockIdx.x * blockDim.x + threadIdx.x;
    if (b > NB) return;
    if (b == NB) { starts[b] = NN; return; }
    int lo = 0, hi = NN;
    while (lo < hi) {
        const int mid = (lo + hi) >> 1;
        if (seg[mid] < b) lo = mid + 1; else hi = mid;
    }
    starts[b] = lo;
}

// ---------------------------------------------------------------------------
// K4: per-bag softmax + weighted pooling. One block per bag, one thread per
//     (p,s) column -> no cross-thread reduction, no atomics, deterministic.
// ---------------------------------------------------------------------------
__global__ __launch_bounds__(128) void pool_kernel(const float* __restrict__ WT,
                                                   const int* __restrict__ starts,
                                                   float* __restrict__ zbag) {
    const int b  = blockIdx.x;
    const int ps = threadIdx.x;
    if (ps >= PS) return;
    const int s0 = starts[b];
    const int s1 = starts[b + 1];
    float m = -3.402823466e38f;
    for (int i = s0; i < s1; ++i)
        m = fmaxf(m, WT[(size_t)i * CC + ps]);
    float denom = 0.f, numer = 0.f;
    for (int i = s0; i < s1; ++i) {
        const float e = __expf(WT[(size_t)i * CC + ps] - m);
        denom += e;
        numer += e * WT[(size_t)i * CC + PS + ps];
    }
    zbag[b * PS + ps] = (s1 > s0) ? (numer / denom) : 0.f;
}

// ---------------------------------------------------------------------------
// K5: per-column (over 4096 bags) mean / unbiased std, scale by b, write out
// ---------------------------------------------------------------------------
__global__ __launch_bounds__(256) void norm_kernel(const float* __restrict__ zbag,
                                                   const float* __restrict__ bvec,
                                                   float* __restrict__ out) {
    __shared__ float sh[256];
    __shared__ float sh2[256];
    const int col = blockIdx.x;   // 0..79
    const int t = threadIdx.x;
    float s = 0.f, s2 = 0.f;
    for (int b = t; b < NB; b += 256) {
        const float v = zbag[b * PS + col];
        s += v; s2 += v * v;
    }
    sh[t] = s; sh2[t] = s2;
    __syncthreads();
    for (int o = 128; o > 0; o >>= 1) {
        if (t < o) { sh[t] += sh[t + o]; sh2[t] += sh2[t + o]; }
        __syncthreads();
    }
    const float sum = sh[0], sumsq = sh2[0];
    const float mean = sum * (1.0f / (float)NB);
    float sd = sqrtf((sumsq - sum * mean) / (float)(NB - 1));
    if (!(sd == sd)) sd = 1.0f;   // reference: NaN std -> 1.0
    const float scale = bvec[col] / sd;
    for (int b = t; b < NB; b += 256) {
        out[b * PS + col] = (zbag[b * PS + col] - mean) * scale;
    }
}

// ---------------------------------------------------------------------------
extern "C" void kernel_launch(void* const* d_in, const int* in_sizes, int n_in,
                              void* d_out, int out_size, void* d_ws, size_t ws_size,
                              hipStream_t stream) {
    const float* x    = (const float*)d_in[0];
    const float* bu   = (const float*)d_in[1];
    const float* bz   = (const float*)d_in[2];
    const int*   seg  = (const int*)d_in[3];
    // d_in[4] = num_segments (constant 4096, baked in)

    char* ws = (char*)d_ws;
    // offsets (256B-aligned)
    half_t* BhT   = (half_t*)(ws + 0);           //  81,920 B
    float*  bvec  = (float*)(ws + 81920);        //     320 B
    float*  WT    = (float*)(ws + 82432);        // 167,772,160 B
    int*    starts= (int*)(ws + 167854592);      //  16,388 B
    float*  zbag  = (float*)(ws + 167871232);    // 1,310,720 B

    float* out = (float*)d_out;                  // [4096, 10, 8] f32

    prep_kernel  <<<1,        256, 0,          stream>>>(bu, bz, BhT, bvec);
    gemm_kernel  <<<NN / 64,  256, SMEM_BYTES, stream>>>(x, BhT, WT);
    bounds_kernel<<<17,       256, 0,          stream>>>(seg, starts);
    pool_kernel  <<<NB,       128, 0,          stream>>>(WT, starts, zbag);
    norm_kernel  <<<PS,       256, 0,          stream>>>(zbag, bvec, out);
}
